// SemanticRegionMaskGuideNorm2d_31078383354208
// MI455X (gfx1250) — matrix operand order repaired and checked by reading, not compiled, not run on hardware
//
#include <hip/hip_runtime.h>
#include <math.h>

typedef __attribute__((ext_vector_type(16))) __bf16 v16bf;
typedef __attribute__((ext_vector_type(8)))  float  v8f;

#define BDIM 8
#define CDIM 64
#define HDIM 256
#define WDIM 256
#define RDIM 19
#define SDIM 64
#define NHDIM 128
#define EPSV 1e-5f

#define PIXT 128                 // pixels per workgroup (one image row segment)
#define COLS (PIXT + 2)          // LDS patch columns incl. halo

// ---------------------------------------------------------------------------
// 1) InstanceNorm statistics over (H,W) per (b,c) on x + eps_noise*noise_var
// ---------------------------------------------------------------------------
__global__ void stats_kernel(const float* __restrict__ x,
                             const float* __restrict__ eps_noise,
                             const float* __restrict__ noise_var,
                             float* __restrict__ stats) {
  int bc = blockIdx.x;                 // b*C + c
  int b = bc / CDIM, c = bc % CDIM;
  const float nv = noise_var[c];
  const float* xp = x + (size_t)bc * (HDIM * WDIM);
  const float* ep = eps_noise + (size_t)b * (HDIM * WDIM);
  float s1 = 0.f, s2 = 0.f;
  for (int i = threadIdx.x; i < HDIM * WDIM; i += 256) {
    float v = xp[i] + ep[i] * nv;
    s1 += v;
    s2 += v * v;
  }
  __shared__ float r1[256], r2[256];
  r1[threadIdx.x] = s1;
  r2[threadIdx.x] = s2;
  __syncthreads();
  for (int off = 128; off > 0; off >>= 1) {
    if (threadIdx.x < off) {
      r1[threadIdx.x] += r1[threadIdx.x + off];
      r2[threadIdx.x] += r2[threadIdx.x + off];
    }
    __syncthreads();
  }
  if (threadIdx.x == 0) {
    const float n = (float)(HDIM * WDIM);
    float mean = r1[0] / n;
    float var  = r2[0] / n - mean * mean;
    stats[bc * 2 + 0] = mean;
    stats[bc * 2 + 1] = rsqrtf(var + EPSV);
  }
}

// ---------------------------------------------------------------------------
// 2) Build style matrix (last-true-region gather), NHWC bf16 for WMMA convs
// ---------------------------------------------------------------------------
__global__ void style_kernel(const int* __restrict__ seg,
                             const float* __restrict__ codes,
                             __bf16* __restrict__ style) {
  size_t pix = (size_t)blockIdx.x * 256 + threadIdx.x;     // over B*H*W (exact)
  int b = (int)(pix / (HDIM * WDIM));
  size_t hw = pix % (HDIM * WDIM);
  const int* sp = seg + (size_t)b * RDIM * (HDIM * WDIM) + hw;
  int j = -1;
  #pragma unroll
  for (int r = 0; r < RDIM; ++r)
    if (sp[(size_t)r * HDIM * WDIM] != 0) j = r;
  __bf16* out = style + pix * SDIM;
  if (j < 0) {
    #pragma unroll 8
    for (int s = 0; s < SDIM; ++s) out[s] = (__bf16)0.f;
  } else {
    const float* cp = codes + ((size_t)b * RDIM + j) * SDIM;
    #pragma unroll 8
    for (int s = 0; s < SDIM; ++s) out[s] = (__bf16)cp[s];
  }
}

// ---------------------------------------------------------------------------
// 3) Pre-swizzle conv weights [64, cin, 3, 3] f32 into bf16 WMMA A-fragments.
//    K ordering: K = (kh*3+kw)*cin + ci.  Fragment layout (contiguous k, as used here):
//    lane<16: M=lane, K={0..15}; lane>=16: M=lane-16, K={16..31}
//    dst index = ((mt*steps + s)*32 + lane)*16 + j
// ---------------------------------------------------------------------------
__global__ void wprep_kernel(const float* __restrict__ w,
                             __bf16* __restrict__ frag, int cin) {
  int steps = 9 * cin / 32;
  int total = 4 * steps * 32 * 16;
  int idx = blockIdx.x * 256 + threadIdx.x;
  if (idx >= total) return;
  int j    = idx & 15;
  int lane = (idx >> 4) & 31;
  int s    = (idx >> 9) % steps;
  int mt   = (idx >> 9) / steps;
  int kk = j + ((lane >= 16) ? 16 : 0);
  int K  = s * 32 + kk;
  int tap = K / cin, ci = K % cin;
  int kh = tap / 3, kw = tap % 3;
  int m = mt * 16 + (lane & 15);
  frag[idx] = (__bf16)w[(((size_t)m * cin + ci) * 3 + kh) * 3 + kw];
}

// ---------------------------------------------------------------------------
// 4) Shared gated conv: mask[B,3,H,W] -> hidden[B,H,W,NH] bf16 with leaky relu
// ---------------------------------------------------------------------------
__global__ __launch_bounds__(256)
void sharedconv_kernel(const float* __restrict__ mask,
                       const float* __restrict__ wsrc,
                       const float* __restrict__ bsrc,
                       __bf16* __restrict__ hidden) {
  __shared__ float wsl[NHDIM * 27];
  __shared__ float bsl[NHDIM];
  for (int i = threadIdx.x; i < NHDIM * 27; i += 256) wsl[i] = wsrc[i];
  if (threadIdx.x < NHDIM) bsl[threadIdx.x] = bsrc[threadIdx.x];
  __syncthreads();
  size_t pix = (size_t)blockIdx.x * 256 + threadIdx.x;   // exact B*H*W grid
  int b = (int)(pix / (HDIM * WDIM));
  int hw = (int)(pix % (HDIM * WDIM));
  int h = hw / WDIM, w = hw % WDIM;
  float m27[27];
  #pragma unroll
  for (int ci = 0; ci < 3; ++ci)
    #pragma unroll
    for (int kh = 0; kh < 3; ++kh)
      #pragma unroll
      for (int kw = 0; kw < 3; ++kw) {
        int hh = h + kh - 1, ww = w + kw - 1;
        float v = 0.f;
        if (hh >= 0 && hh < HDIM && ww >= 0 && ww < WDIM)
          v = mask[(((size_t)b * 3 + ci) * HDIM + hh) * WDIM + ww];
        m27[(ci * 3 + kh) * 3 + kw] = v;
      }
  __bf16* out = hidden + pix * NHDIM;
  for (int co = 0; co < NHDIM; ++co) {
    float acc = bsl[co];
    #pragma unroll
    for (int t = 0; t < 27; ++t) acc = fmaf(wsl[co * 27 + t], m27[t], acc);
    acc = acc > 0.f ? acc : 0.2f * acc;
    out[co] = (__bf16)acc;
  }
}

// ---------------------------------------------------------------------------
// Fused WMMA conv helpers
// ---------------------------------------------------------------------------
template <int CIN>
__device__ __forceinline__ void fill_patch(__bf16* __restrict__ patch,
                                           const __bf16* __restrict__ in,
                                           int b, int h, int wbase) {
  constexpr int NCHUNK = 3 * COLS * CIN / 8;     // 16B chunks
  for (int chunk = threadIdx.x; chunk < NCHUNK; chunk += 256) {
    int e0 = chunk * 8;
    int ci  = e0 % CIN;
    int rc  = e0 / CIN;
    int col = rc % COLS, row = rc / COLS;
    int hh = h + row - 1, ww = wbase + col - 1;
    uint4 v = make_uint4(0u, 0u, 0u, 0u);
    if (hh >= 0 && hh < HDIM && ww >= 0 && ww < WDIM)
      v = *reinterpret_cast<const uint4*>(in + (((size_t)b * HDIM + hh) * WDIM + ww) * CIN + ci);
    *reinterpret_cast<uint4*>(patch + e0) = v;
  }
}

template <int CIN>
__device__ __forceinline__ void conv_phase(const __bf16* __restrict__ patch,
                                           const v16bf* __restrict__ aG,   // pre-offset by mt,lane
                                           const v16bf* __restrict__ aB,
                                           int pcol0, int hi16,
                                           v8f accG[4], v8f accB[4]) {
  constexpr int STEPS = 9 * CIN / 32;
  constexpr int CH    = CIN / 32;
  for (int s = 0; s < STEPS; ++s) {
    const int tap = s / CH;
    const int chunk = (s % CH) * 32;
    const int kh = tap / 3, kw = tap % 3;
    v16bf ag = aG[s * 32];
    v16bf ab = aB[s * 32];
    #pragma unroll
    for (int n = 0; n < 4; ++n) {
      v16bf bf = *reinterpret_cast<const v16bf*>(
          patch + ((kh * COLS + pcol0 + n * 16 + kw) * CIN + chunk + hi16));
      accG[n] = __builtin_amdgcn_wmma_f32_16x16x32_bf16(false, ag, false, bf,
                                                        (short)0, accG[n], false, false);
      accB[n] = __builtin_amdgcn_wmma_f32_16x16x32_bf16(false, ab, false, bf,
                                                        (short)0, accB[n], false, false);
    }
  }
}

// ---------------------------------------------------------------------------
// 5) Fused implicit-GEMM: both 3x3 convs (style 64->64 and gated 128->64) for
//    64 out-channels x 128 pixels per workgroup. 8 waves = 4 m-tiles x 2
//    pixel-groups; each wave owns 4 n-tiles (64 px) so every A fragment feeds
//    4 WMMAs. Epilogue blends sigmoid(g)*style + (1-sigmoid(g))*gated once.
// ---------------------------------------------------------------------------
__global__ __launch_bounds__(256)
void conv_fused_kernel(const __bf16* __restrict__ style,   // [B,H,W,64]
                       const __bf16* __restrict__ hidden,  // [B,H,W,128]
                       const __bf16* __restrict__ fragSg,  // style gamma A frags
                       const __bf16* __restrict__ fragSb,  // style beta  A frags
                       const __bf16* __restrict__ fragGg,  // gated gamma A frags
                       const __bf16* __restrict__ fragGb,  // gated beta  A frags
                       const float* __restrict__ biasSg, const float* __restrict__ biasSb,
                       const float* __restrict__ biasGg, const float* __restrict__ biasGb,
                       const float* __restrict__ blendG, const float* __restrict__ blendB,
                       float* __restrict__ outG,           // [B,C,H,W] f32
                       float* __restrict__ outB) {
  constexpr int STEPS_S = 9 * SDIM / 32;    // 18
  constexpr int STEPS_H = 9 * NHDIM / 32;   // 36
  alignas(32) __shared__ __bf16 patchS[3 * COLS * SDIM];    // ~49 KB
  alignas(32) __shared__ __bf16 patchH[3 * COLS * NHDIM];   // ~98 KB

  int blk = blockIdx.x;                 // B * H * (W/PIXT) = 4096
  int wt = blk & 1;
  int h  = (blk >> 1) & (HDIM - 1);
  int b  = blk >> 9;
  int wbase = wt * PIXT;

  fill_patch<SDIM>(patchS, style, b, h, wbase);
  fill_patch<NHDIM>(patchH, hidden, b, h, wbase);
  __syncthreads();

  int lane = threadIdx.x & 31;
  int wave = threadIdx.x >> 5;
  int mt = wave & 3;                    // 16 out-channels
  int pg = wave >> 2;                   // pixel group (64 px)
  int pcol0 = pg * 64 + (lane & 15);
  int hi = (lane >> 4) & 1;
  int hi16 = hi * 16;

  const v16bf* aSg = reinterpret_cast<const v16bf*>(fragSg) + (size_t)mt * STEPS_S * 32 + lane;
  const v16bf* aSb = reinterpret_cast<const v16bf*>(fragSb) + (size_t)mt * STEPS_S * 32 + lane;
  const v16bf* aGg = reinterpret_cast<const v16bf*>(fragGg) + (size_t)mt * STEPS_H * 32 + lane;
  const v16bf* aGb = reinterpret_cast<const v16bf*>(fragGb) + (size_t)mt * STEPS_H * 32 + lane;

  v8f sG[4] = {}, sB[4] = {}, gG[4] = {}, gB[4] = {};
  conv_phase<SDIM>(patchS, aSg, aSb, pcol0, hi16, sG, sB);
  conv_phase<NHDIM>(patchH, aGg, aGb, pcol0, hi16, gG, gB);

  float ga = 1.f / (1.f + __expf(-blendG[0]));
  float ba = 1.f / (1.f + __expf(-blendB[0]));
  #pragma unroll
  for (int n = 0; n < 4; ++n) {
    int pw = wbase + pcol0 + n * 16;
    #pragma unroll
    for (int r = 0; r < 8; ++r) {
      int co = mt * 16 + r + hi * 8;    // D layout: lanes>=16 hold M = r+8
      size_t addr = (((size_t)b * CDIM + co) * HDIM + h) * WDIM + pw;
      outG[addr] = ga * (sG[n][r] + biasSg[co]) + (1.f - ga) * (gG[n][r] + biasGg[co]);
      outB[addr] = ba * (sB[n][r] + biasSb[co]) + (1.f - ba) * (gB[n][r] + biasGb[co]);
    }
  }
}

// ---------------------------------------------------------------------------
// 6) Finalize: instance-norm + affine + (1+gamma)*x + beta, NCHW elementwise
// ---------------------------------------------------------------------------
__global__ void finalize_kernel(const float* __restrict__ x,
                                const float* __restrict__ eps_noise,
                                const float* __restrict__ noise_var,
                                const float* __restrict__ in_w,
                                const float* __restrict__ in_b,
                                const float* __restrict__ stats,
                                const float* __restrict__ gammaF,
                                const float* __restrict__ betaF,
                                float* __restrict__ out) {
  size_t idx = (size_t)blockIdx.x * 256 + threadIdx.x;    // exact grid
  size_t hw = idx % (HDIM * WDIM);
  int c = (int)((idx / (HDIM * WDIM)) % CDIM);
  int b = (int)(idx / ((size_t)CDIM * HDIM * WDIM));
  float xv = x[idx] + eps_noise[(size_t)b * HDIM * WDIM + hw] * noise_var[c];
  float mean = stats[(b * CDIM + c) * 2 + 0];
  float rstd = stats[(b * CDIM + c) * 2 + 1];
  float xn = (xv - mean) * rstd * in_w[c] + in_b[c];
  out[idx] = xn * (1.f + gammaF[idx]) + betaF[idx];
}

// ---------------------------------------------------------------------------
extern "C" void kernel_launch(void* const* d_in, const int* in_sizes, int n_in,
                              void* d_out, int out_size, void* d_ws, size_t ws_size,
                              hipStream_t stream) {
  const float* x         = (const float*)d_in[0];
  const int*   seg       = (const int*)d_in[1];
  const float* codes     = (const float*)d_in[2];
  const float* mask      = (const float*)d_in[3];
  const float* eps_noise = (const float*)d_in[4];
  const float* noise_var = (const float*)d_in[5];
  const float* in_w      = (const float*)d_in[6];
  const float* in_b      = (const float*)d_in[7];
  const float* w_gamma   = (const float*)d_in[8];
  const float* b_gamma   = (const float*)d_in[9];
  const float* w_beta    = (const float*)d_in[10];
  const float* b_beta    = (const float*)d_in[11];
  const float* w_shared  = (const float*)d_in[12];
  const float* b_shared  = (const float*)d_in[13];
  const float* w_ggamma  = (const float*)d_in[14];
  const float* b_ggamma  = (const float*)d_in[15];
  const float* w_gbeta   = (const float*)d_in[16];
  const float* b_gbeta   = (const float*)d_in[17];
  const float* blend_g   = (const float*)d_in[18];
  const float* blend_b   = (const float*)d_in[19];
  float* out = (float*)d_out;

  // workspace carve-up (256B aligned regions)
  char* ws = (char*)d_ws;
  const size_t NPIX   = (size_t)BDIM * HDIM * WDIM;              // 524288
  const size_t NELEM  = NPIX * CDIM;                             // 33.5M
  size_t off = 0;
  auto carve = [&](size_t bytes) { size_t o = off; off += (bytes + 255) & ~(size_t)255; return o; };
  float*  stats   = (float*)(ws + carve((size_t)BDIM * CDIM * 2 * sizeof(float)));
  __bf16* style   = (__bf16*)(ws + carve(NPIX * SDIM * sizeof(__bf16)));
  __bf16* hidden  = (__bf16*)(ws + carve(NPIX * NHDIM * sizeof(__bf16)));
  float*  gammaF  = (float*)(ws + carve(NELEM * sizeof(float)));
  float*  betaF   = (float*)(ws + carve(NELEM * sizeof(float)));
  const int frag64  = 4 * (9 * SDIM / 32) * 32 * 16;    // 36864 elems
  const int frag128 = 4 * (9 * NHDIM / 32) * 32 * 16;   // 73728 elems
  __bf16* fragWg = (__bf16*)(ws + carve((size_t)frag64 * sizeof(__bf16)));
  __bf16* fragWb = (__bf16*)(ws + carve((size_t)frag64 * sizeof(__bf16)));
  __bf16* fragGg = (__bf16*)(ws + carve((size_t)frag128 * sizeof(__bf16)));
  __bf16* fragGb = (__bf16*)(ws + carve((size_t)frag128 * sizeof(__bf16)));

  // 1) instance-norm stats
  stats_kernel<<<BDIM * CDIM, 256, 0, stream>>>(x, eps_noise, noise_var, stats);

  // 2) style gather -> NHWC bf16
  style_kernel<<<(int)(NPIX / 256), 256, 0, stream>>>(seg, codes, style);

  // 3) weight fragment swizzles
  wprep_kernel<<<(frag64 + 255) / 256, 256, 0, stream>>>(w_gamma, fragWg, SDIM);
  wprep_kernel<<<(frag64 + 255) / 256, 256, 0, stream>>>(w_beta, fragWb, SDIM);
  wprep_kernel<<<(frag128 + 255) / 256, 256, 0, stream>>>(w_ggamma, fragGg, NHDIM);
  wprep_kernel<<<(frag128 + 255) / 256, 256, 0, stream>>>(w_gbeta, fragGb, NHDIM);

  // 4) shared conv + leaky relu -> hidden NHWC bf16
  sharedconv_kernel<<<(int)(NPIX / 256), 256, 0, stream>>>(mask, w_shared, b_shared, hidden);

  // 5) fused WMMA convs + blending epilogue
  const int convGrid = BDIM * HDIM * (WDIM / PIXT);   // 4096
  conv_fused_kernel<<<convGrid, 256, 0, stream>>>(
      style, hidden, fragWg, fragWb, fragGg, fragGb,
      b_gamma, b_beta, b_ggamma, b_gbeta, blend_g, blend_b, gammaF, betaF);

  // 6) finalize
  finalize_kernel<<<(int)(NELEM / 256), 256, 0, stream>>>(
      x, eps_noise, noise_var, in_w, in_b, stats, gammaF, betaF, out);
}
